// DIMBA_35450660062042
// MI455X (gfx1250) — matrix operand order repaired and checked by reading, not compiled, not run on hardware
//
#include <hip/hip_runtime.h>
#include <hip/hip_bf16.h>
#include <math.h>

// ---------------- problem constants ----------------
#define BATCH 8
#define SEQ   1024
#define NTOK  (BATCH*SEQ)          // 8192 rows
#define DMODEL 768
#define DPROJ  768
#define NLAYER 6
#define DSTATE 16
#define DCONV  4
#define DINNER 1536
#define DTRANK 48
#define TIME_DIM 128
#define TIME_OUT 512
#define T_STEPS 1000

typedef __attribute__((ext_vector_type(16))) __bf16 v16bf;
typedef __attribute__((ext_vector_type(8)))  float  v8f;

#define PI_2F 1.57079632679489662f

// ---------------- small device helpers ----------------
__device__ __forceinline__ unsigned short f2bf(float f) {
  unsigned u = __float_as_uint(f);
  u += 0x7FFFu + ((u >> 16) & 1u);       // round-to-nearest-even
  return (unsigned short)(u >> 16);
}
__device__ __forceinline__ unsigned pack2bf(float lo, float hi) {
#if __has_builtin(__builtin_amdgcn_cvt_pk_bf16_f32)
  typedef __attribute__((ext_vector_type(2))) __bf16 v2bf;
  v2bf r = __builtin_amdgcn_cvt_pk_bf16_f32(lo, hi);
  unsigned u; __builtin_memcpy(&u, &r, 4);
  return u;
#else
  return (unsigned)f2bf(lo) | ((unsigned)f2bf(hi) << 16);
#endif
}
__device__ __forceinline__ float sigmoid_(float x){ return 1.f/(1.f+__expf(-x)); }
__device__ __forceinline__ float silu_(float x){ return x*sigmoid_(x); }
__device__ __forceinline__ float gelu_(float x){
  float x3 = x*x*x;
  return 0.5f*x*(1.f + tanhf(0.7978845608028654f*(x + 0.044715f*x3)));
}
__device__ __forceinline__ float softplus_(float x){
  return (x > 20.f) ? x : log1pf(__expf(x));
}

// ---------------- WMMA GEMM: C[M,N] = epi(A[M,K] * B[K,N] + bias) ----------------
// block = 256 threads (8 waves). Block tile 128(M) x 128(N); wave tile 32x64
// (2x4 WMMA); K-step 32; bf16 operands in ping-pong LDS; f32 accumulate.
#define EPI_NONE 0
#define EPI_GELU 1
#define EPI_SOFTPLUS 2
#define EPI_ADD 3

template<int MODE>
__global__ __launch_bounds__(256)
void gemm_wmma(const float* __restrict__ A, const float* __restrict__ Bw,
               const float* __restrict__ bias, float* __restrict__ C,
               int M, int N, int K)
{
  __shared__ unsigned short As[2][128][40];   // [buf][m][k], row stride 80B
  __shared__ unsigned short Bs[2][128][40];   // [buf][n][k] (transposed store)

  const int tid  = threadIdx.x;
  const int wave = tid >> 5;
  const int lane = tid & 31;
  const int wr = wave >> 1, wc = wave & 1;   // 4x2 wave grid -> 32x64 wave tile
  const int g  = lane >> 4, ln = lane & 15;  // half-wave group / lane-in-group
  const int m0 = blockIdx.y * 128;
  const int n0 = blockIdx.x * 128;

  const bool mnfull = (m0 + 128 <= M) && (n0 + 128 <= N);  // block-uniform

  v8f zero = {};
  v8f acc[2][4];
  #pragma unroll
  for (int i = 0; i < 2; i++)
    #pragma unroll
    for (int j = 0; j < 4; j++) acc[i][j] = zero;

  auto stage = [&](int pb, int kc) {
    if (mnfull && (kc + 32 <= K)) {
      // ---- fast path: unguarded vector loads, packed b64 LDS stores ----
      #pragma unroll
      for (int i = 0; i < 4; i++) {          // A: 128x32 = 1024 float4 groups
        int vid = tid + i*256;
        int r = vid >> 3, kq = (vid & 7) << 2;
        const float4 f = *(const float4*)(A + (long)(m0 + r)*K + kc + kq);
        uint2 pk; pk.x = pack2bf(f.x, f.y); pk.y = pack2bf(f.z, f.w);
        *(uint2*)&As[pb][r][kq] = pk;
      }
      #pragma unroll
      for (int i = 0; i < 4; i++) {          // B: 32x128, (n, 4k) groups
        int vid = tid + i*256;
        int n = vid & 127, kq = (vid >> 7) << 2;
        long base = (long)(kc + kq)*N + n0 + n;
        float b0 = Bw[base];
        float b1 = Bw[base + (long)N];
        float b2 = Bw[base + 2L*N];
        float b3 = Bw[base + 3L*N];
        uint2 pk; pk.x = pack2bf(b0, b1); pk.y = pack2bf(b2, b3);
        *(uint2*)&Bs[pb][n][kq] = pk;
      }
      if (kc + 64 <= K) {                    // prefetch tile staged next iteration
        __builtin_prefetch(A + (long)(m0 + (tid >> 1))*K + kc + 32 + (tid & 1)*16, 0, 3);
        __builtin_prefetch(Bw + (long)(kc + 32 + (tid >> 3))*N + n0 + (tid & 7)*16, 0, 3);
      }
    } else {
      // ---- slow path (edge tiles only: small-N GEMMs, K=48 tail) ----
      #pragma unroll
      for (int i = 0; i < 16; i++) {
        int idx = tid + i*256;
        int r = idx >> 5, k = idx & 31;
        int gr = m0 + r, gk = kc + k;
        float v = (gr < M && gk < K) ? A[(long)gr*K + gk] : 0.f;
        As[pb][r][k] = f2bf(v);
      }
      #pragma unroll
      for (int i = 0; i < 16; i++) {
        int idx = tid + i*256;
        int k = idx >> 7, n = idx & 127;
        int gk = kc + k, gn = n0 + n;
        float v = (gk < K && gn < N) ? Bw[(long)gk*N + gn] : 0.f;
        Bs[pb][n][k] = f2bf(v);
      }
    }
  };

  stage(0, 0);
  __syncthreads();
  int pb = 0;
  for (int k0 = 0; k0 < K; k0 += 32) {
    if (k0 + 32 < K) stage(pb ^ 1, k0 + 32);   // overlap next-tile loads with WMMA

    union FU { unsigned int u[8]; v16bf v; };
    FU a[2], b[4];
    // A fragment: 16-bit A 16x32 layout (ISA 7.12.2)
    #pragma unroll
    for (int mi = 0; mi < 2; mi++) {
      int m = wr*32 + mi*16 + ln;
      #pragma unroll
      for (int j = 0; j < 8; j++) {
        int kb = ((j >> 2) << 4) + (g << 3) + ((j & 3) << 1);
        a[mi].u[j] = *(const unsigned int*)&As[pb][m][kb];
      }
    }
    // B fragment: same K order as A (16-bit A 16x32 layout), 2 K per VGPR
    #pragma unroll
    for (int ni = 0; ni < 4; ni++) {
      int n = wc*64 + ni*16 + ln;
      #pragma unroll
      for (int j = 0; j < 8; j++) {
        int kb = ((j >> 2) << 4) + (g << 3) + ((j & 3) << 1);
        b[ni].u[j] = *(const unsigned int*)&Bs[pb][n][kb];
      }
    }
    #pragma unroll
    for (int mi = 0; mi < 2; mi++)
      #pragma unroll
      for (int ni = 0; ni < 4; ni++)
        acc[mi][ni] = __builtin_amdgcn_wmma_f32_16x16x32_bf16(
            false, a[mi].v, false, b[ni].v, (short)0, acc[mi][ni], false, false);
    __syncthreads();   // single barrier per K-step (ping-pong)
    pb ^= 1;
  }

  // --- epilogue: C/D layout VGPR r -> M = g*8 + r, N = ln ---
  #pragma unroll
  for (int mi = 0; mi < 2; mi++) {
    #pragma unroll
    for (int ni = 0; ni < 4; ni++) {
      #pragma unroll
      for (int r = 0; r < 8; r++) {
        int m = m0 + wr*32 + mi*16 + g*8 + r;
        int n = n0 + wc*64 + ni*16 + ln;
        if (m < M && n < N) {
          float v = acc[mi][ni][r];
          if (bias) v += bias[n];
          if (MODE == EPI_GELU)     v = gelu_(v);
          if (MODE == EPI_SOFTPLUS) v = softplus_(v);
          long idx = (long)m*N + n;
          if (MODE == EPI_ADD) v += C[idx];
          C[idx] = v;
        }
      }
    }
  }
}

// ---------------- embedding + cosine-schedule noising ----------------
__global__ void embed_kernel(const int* __restrict__ ids, const int* __restrict__ t,
                             const float* __restrict__ noise, const float* __restrict__ E,
                             float* __restrict__ x0, float* __restrict__ x, long total)
{
  long i = (long)blockIdx.x*blockDim.x + threadIdx.x;
  if (i >= total) return;
  int col = (int)(i % DMODEL);
  long row = i / DMODEL;
  int b = (int)(row / SEQ);
  int tok = ids[row];
  float v0 = E[(long)tok*DMODEL + col];
  const float sC = 0.008f;
  float c0 = cosf(sC/(1.f+sC) * PI_2F);
  float f0 = c0*c0;
  float step = (float)(t[b] + 1);
  float cc = cosf(((step/(float)T_STEPS) + sC)/(1.f+sC) * PI_2F);
  float ac = cc*cc / f0;
  ac = fminf(fmaxf(ac, 1e-8f), 1.f);
  x0[i] = v0;
  x[i]  = sqrtf(ac)*v0 + sqrtf(1.f-ac)*noise[i];
}

// ---------------- timestep embedding MLP (tiny: per-sample block) ----------------
__global__ __launch_bounds__(512)
void temb_kernel(const int* __restrict__ t, const float* __restrict__ W1, const float* __restrict__ b1,
                 const float* __restrict__ W2, const float* __restrict__ b2, float* __restrict__ temb)
{
  __shared__ float emb[TIME_DIM];
  __shared__ float hid[TIME_OUT];
  int b = blockIdx.x;
  int j = threadIdx.x;
  if (j < TIME_DIM/2) {
    float fr = __expf(-logf(10000.f) * (float)j / (float)(TIME_DIM/2));
    float ang = (float)t[b] * fr;
    emb[j] = sinf(ang);
    emb[j + TIME_DIM/2] = cosf(ang);
  }
  __syncthreads();
  float h = b1[j];
  for (int i = 0; i < TIME_DIM; i++) h += emb[i]*W1[i*TIME_OUT + j];
  hid[j] = silu_(h);
  __syncthreads();
  float o = b2[j];
  for (int i = 0; i < TIME_OUT; i++) o += hid[i]*W2[i*TIME_OUT + j];
  temb[b*TIME_OUT + j] = o;
}

// ---------------- temb @ film_Wt[l] : [B,512]x[512,1536] ----------------
__global__ void tfilm_kernel(const float* __restrict__ temb, const float* __restrict__ Wt,
                             float* __restrict__ tf)
{
  int b = blockIdx.y;
  int c = blockIdx.x*blockDim.x + threadIdx.x;   // < 2*DMODEL
  float acc = 0.f;
  for (int i = 0; i < TIME_OUT; i++) acc += temb[b*TIME_OUT + i] * Wt[(long)i*(2*DMODEL) + c];
  tf[b*(2*DMODEL) + c] = acc;
}

// ---------------- LayerNorm (block per row, D=768) ----------------
__global__ __launch_bounds__(256)
void ln_kernel(const float* __restrict__ x, const float* __restrict__ scale,
               const float* __restrict__ bias, float* __restrict__ out)
{
  __shared__ float red[256];
  int row = blockIdx.x;
  int tid = threadIdx.x;
  const float* xr = x + (long)row*DMODEL;
  float s = 0.f;
  for (int i = tid; i < DMODEL; i += 256) s += xr[i];
  red[tid] = s; __syncthreads();
  for (int k = 128; k > 0; k >>= 1){ if (tid < k) red[tid] += red[tid+k]; __syncthreads(); }
  float mean = red[0] / (float)DMODEL;
  __syncthreads();
  float v = 0.f;
  for (int i = tid; i < DMODEL; i += 256){ float d = xr[i]-mean; v += d*d; }
  red[tid] = v; __syncthreads();
  for (int k = 128; k > 0; k >>= 1){ if (tid < k) red[tid] += red[tid+k]; __syncthreads(); }
  float rstd = rsqrtf(red[0]/(float)DMODEL + 1e-5f);
  for (int i = tid; i < DMODEL; i += 256)
    out[(long)row*DMODEL + i] = scale[i]*(xr[i]-mean)*rstd + bias[i];
}

// ---------------- FiLM apply: h = (1+gamma)*h + beta ----------------
__global__ void film_apply(const float* __restrict__ film, const float* __restrict__ tf,
                           float* __restrict__ h, long total)
{
  long i = (long)blockIdx.x*blockDim.x + threadIdx.x;
  if (i >= total) return;
  int col = (int)(i % DMODEL);
  long row = i / DMODEL;
  int b = (int)(row / SEQ);
  const float* fr = film + row*(2*DMODEL);
  const float* tr = tf + (long)b*(2*DMODEL);
  float gamma = fr[col] + tr[col];
  float beta  = fr[DMODEL+col] + tr[DMODEL+col];
  h[i] = (1.f + gamma)*h[i] + beta;
}

// ---------------- depthwise causal conv (DK=4) + SiLU ----------------
__global__ void conv_silu(const float* __restrict__ xz, const float* __restrict__ cw,
                          const float* __restrict__ cb, float* __restrict__ xc, long total)
{
  long i = (long)blockIdx.x*blockDim.x + threadIdx.x;
  if (i >= total) return;
  int c = (int)(i % DINNER);
  long row = i / DINNER;
  int s = (int)(row % SEQ);
  float acc = cb[c];
  #pragma unroll
  for (int k = 0; k < DCONV; k++) {
    int sp = s + k - (DCONV-1);
    if (sp >= 0)
      acc += xz[(row + (long)(k - (DCONV-1)))*(2*DINNER) + c] * cw[c*DCONV + k];
  }
  xc[i] = silu_(acc);
}

// ---------------- pack W_B|W_C into one [DINNER,32] weight ----------------
__global__ void pack_bc_kernel(const float* __restrict__ WB, const float* __restrict__ WC,
                               float* __restrict__ WBC)
{
  int i = blockIdx.x*256 + threadIdx.x;   // < DINNER*DSTATE
  if (i >= DINNER*DSTATE) return;
  int r = i / DSTATE, c = i % DSTATE;
  WBC[r*32 + c]          = WB[i];
  WBC[r*32 + DSTATE + c] = WC[i];
}

// ---------------- selective scan: thread per (batch, channel) ----------------
// BC holds [NTOK,32]: cols 0..15 = B_t, cols 16..31 = C_t.
__global__ __launch_bounds__(256)
void scan_kernel(const float* __restrict__ xc, const float* __restrict__ dt,
                 const float* __restrict__ BC, const float* __restrict__ Al,
                 float* __restrict__ ys)
{
  int b = blockIdx.y;
  int d = blockIdx.x*256 + threadIdx.x;     // < DINNER
  float A[DSTATE], h[DSTATE];
  #pragma unroll
  for (int j = 0; j < DSTATE; j++) { A[j] = -__expf(Al[d*DSTATE + j]); h[j] = 0.f; }
  const float* xcb = xc + (long)b*SEQ*DINNER;
  const float* dtb = dt + (long)b*SEQ*DINNER;
  const float* bcb = BC + (long)b*SEQ*32;
  float* yb = ys + (long)b*SEQ*DINNER;
  for (int t = 0; t < SEQ; t++) {
    float xt  = xcb[(long)t*DINNER + d];
    float dtt = dtb[(long)t*DINNER + d];
    float dx = dtt * xt;
    float y = 0.f;
    #pragma unroll
    for (int j = 0; j < DSTATE; j++) {
      float dA = __expf(dtt * A[j]);
      h[j] = dA*h[j] + dx*bcb[t*32 + j];
      y += h[j]*bcb[t*32 + DSTATE + j];
    }
    yb[(long)t*DINNER + d] = y;
  }
}

// ---------------- gating: y = (ys + xc*Dp) * silu(z) ----------------
__global__ void gate_kernel(const float* __restrict__ xc, const float* __restrict__ xz,
                            const float* __restrict__ Dp, float* __restrict__ ys, long total)
{
  long i = (long)blockIdx.x*blockDim.x + threadIdx.x;
  if (i >= total) return;
  int c = (int)(i % DINNER);
  long row = i / DINNER;
  float z = xz[row*(2*DINNER) + DINNER + c];
  ys[i] = (ys[i] + xc[i]*Dp[c]) * silu_(z);
}

// ---------------- host side ----------------
static void launch_gemm(int mode, const float* A, const float* Bw, const float* bias, float* C,
                        int M, int N, int K, hipStream_t stream) {
  dim3 grid((N + 127)/128, (M + 127)/128), blk(256);
  if      (mode == EPI_NONE)     gemm_wmma<EPI_NONE><<<grid, blk, 0, stream>>>(A, Bw, bias, C, M, N, K);
  else if (mode == EPI_GELU)     gemm_wmma<EPI_GELU><<<grid, blk, 0, stream>>>(A, Bw, bias, C, M, N, K);
  else if (mode == EPI_SOFTPLUS) gemm_wmma<EPI_SOFTPLUS><<<grid, blk, 0, stream>>>(A, Bw, bias, C, M, N, K);
  else                           gemm_wmma<EPI_ADD><<<grid, blk, 0, stream>>>(A, Bw, bias, C, M, N, K);
}

extern "C" void kernel_launch(void* const* d_in, const int* in_sizes, int n_in,
                              void* d_out, int out_size, void* d_ws, size_t ws_size,
                              hipStream_t stream) {
  const int*   ids      = (const int*)  d_in[0];
  const int*   t_in     = (const int*)  d_in[1];
  const float* noise    = (const float*)d_in[2];
  const float* E        = (const float*)d_in[3];
  const float* pe_W1    = (const float*)d_in[4];
  const float* pe_b1    = (const float*)d_in[5];
  const float* pe_W2    = (const float*)d_in[6];
  const float* pe_b2    = (const float*)d_in[7];
  const float* te_W1    = (const float*)d_in[8];
  const float* te_b1    = (const float*)d_in[9];
  const float* te_W2    = (const float*)d_in[10];
  const float* te_b2    = (const float*)d_in[11];
  const float* ln_scale = (const float*)d_in[12];
  const float* ln_bias  = (const float*)d_in[13];
  const float* film_Wc  = (const float*)d_in[14];
  const float* film_Wt  = (const float*)d_in[15];
  const float* film_b   = (const float*)d_in[16];
  const float* W_in     = (const float*)d_in[17];
  const float* conv_w   = (const float*)d_in[18];
  const float* conv_b   = (const float*)d_in[19];
  const float* W_xdt    = (const float*)d_in[20];
  const float* W_dt     = (const float*)d_in[21];
  const float* dt_bias  = (const float*)d_in[22];
  const float* W_B      = (const float*)d_in[23];
  const float* W_C      = (const float*)d_in[24];
  const float* A_log    = (const float*)d_in[25];
  const float* D_p      = (const float*)d_in[26];
  const float* W_out    = (const float*)d_in[27];
  const float* b_out    = (const float*)d_in[28];
  const float* lnf_s    = (const float*)d_in[29];
  const float* lnf_b    = (const float*)d_in[30];

  const long totD  = (long)NTOK*DMODEL;
  const long totDI = (long)NTOK*DINNER;

  // workspace partition (floats)
  float* ws = (float*)d_ws;
  long off = 0;
  float* x0    = ws + off; off += totD;
  float* x     = ws + off; off += totD;
  float* cond  = ws + off; off += (long)NTOK*DPROJ;
  float* h     = ws + off; off += totD;
  float* film  = ws + off; off += (long)NTOK*2*DMODEL;  // also cond1 scratch
  float* xz    = ws + off; off += (long)NTOK*2*DINNER;
  float* xc    = ws + off; off += totDI;
  float* dt    = ws + off; off += totDI;
  float* ysb   = ws + off; off += totDI;
  float* dtlow = ws + off; off += (long)NTOK*DTRANK;
  float* BCm   = ws + off; off += (long)NTOK*32;
  float* WBC   = ws + off; off += (long)DINNER*32;
  float* temb  = ws + off; off += (long)BATCH*TIME_OUT;
  float* tf    = ws + off; off += (long)BATCH*2*DMODEL;

  // 1) embedding + noising
  embed_kernel<<<(unsigned)((totD + 255)/256), 256, 0, stream>>>(ids, t_in, noise, E, x0, x, totD);

  // 2) prompt encoder: cond = gelu(x0@pe_W1+b1) @ pe_W2 + b2
  launch_gemm(EPI_GELU, x0, pe_W1, pe_b1, film, NTOK, 2*DMODEL, DMODEL, stream);
  launch_gemm(EPI_NONE, film, pe_W2, pe_b2, cond, NTOK, DPROJ, 2*DMODEL, stream);

  // 3) timestep embedding
  temb_kernel<<<BATCH, TIME_OUT, 0, stream>>>(t_in, te_W1, te_b1, te_W2, te_b2, temb);

  for (int l = 0; l < NLAYER; l++) {
    ln_kernel<<<NTOK, 256, 0, stream>>>(x, ln_scale + l*DMODEL, ln_bias + l*DMODEL, h);
    tfilm_kernel<<<dim3((2*DMODEL)/256, BATCH), 256, 0, stream>>>(
        temb, film_Wt + (long)l*TIME_OUT*2*DMODEL, tf);
    launch_gemm(EPI_NONE, cond, film_Wc + (long)l*DPROJ*2*DMODEL, film_b + l*2*DMODEL,
                film, NTOK, 2*DMODEL, DPROJ, stream);
    film_apply<<<(unsigned)((totD + 255)/256), 256, 0, stream>>>(film, tf, h, totD);
    launch_gemm(EPI_NONE, h, W_in + (long)l*DMODEL*2*DINNER, nullptr, xz, NTOK, 2*DINNER, DMODEL, stream);
    conv_silu<<<(unsigned)((totDI + 255)/256), 256, 0, stream>>>(
        xz, conv_w + (long)l*DINNER*DCONV, conv_b + l*DINNER, xc, totDI);
    launch_gemm(EPI_NONE, xc, W_xdt + (long)l*DINNER*DTRANK, nullptr, dtlow, NTOK, DTRANK, DINNER, stream);
    launch_gemm(EPI_SOFTPLUS, dtlow, W_dt + (long)l*DTRANK*DINNER, dt_bias + l*DINNER,
                dt, NTOK, DINNER, DTRANK, stream);
    pack_bc_kernel<<<(DINNER*DSTATE + 255)/256, 256, 0, stream>>>(
        W_B + (long)l*DINNER*DSTATE, W_C + (long)l*DINNER*DSTATE, WBC);
    launch_gemm(EPI_NONE, xc, WBC, nullptr, BCm, NTOK, 32, DINNER, stream);
    scan_kernel<<<dim3(DINNER/256, BATCH), 256, 0, stream>>>(
        xc, dt, BCm, A_log + (long)l*DINNER*DSTATE, ysb);
    gate_kernel<<<(unsigned)((totDI + 255)/256), 256, 0, stream>>>(xc, xz, D_p + l*DINNER, ysb, totDI);
    launch_gemm(EPI_ADD, ysb, W_out + (long)l*DINNER*DMODEL, b_out + l*DMODEL, x, NTOK, DMODEL, DINNER, stream);
  }

  // final LN -> first output; copy noise -> second output
  float* outp = (float*)d_out;
  ln_kernel<<<NTOK, 256, 0, stream>>>(x, lnf_s, lnf_b, outp);
  hipMemcpyAsync(outp + totD, noise, totD*sizeof(float), hipMemcpyDeviceToDevice, stream);
}
